// SwinCrossAttention_52647709114607
// MI455X (gfx1250) — matrix operand order repaired and checked by reading, not compiled, not run on hardware
//
#include <hip/hip_runtime.h>
#include <hip/hip_bf16.h>
#include <cstdint>

typedef __attribute__((ext_vector_type(16))) _Float16 v16h;
typedef __attribute__((ext_vector_type(8)))  float    v8f;
typedef __attribute__((ext_vector_type(8)))  unsigned v8u;

#define DEV __device__ __forceinline__

// ---------------------------------------------------------------------------
// WMMA helpers (CDNA5 gfx1250, wave32). D = A(16x32 f16) * B(32x16 f16) + C(f32)
// ---------------------------------------------------------------------------
DEV v8f wmma_f16(v16h a, v16h b, v8f c) {
  return __builtin_amdgcn_wmma_f32_16x16x32_f16(
      /*neg_a=*/false, a, /*neg_b=*/false, b,
      /*c_mod=*/(short)0, c, /*reuse_a=*/false, /*reuse_b=*/false);
}

// A fragment (16x32, MxK) from row-major f16 source, leading dim ld (even).
// Lanes 0-15: M=lane, K0..15 contiguous; lanes 16-31: K16..31.
DEV v16h load_a32(const _Float16* p, int ld) {
  int lane = threadIdx.x & 31;
  int m    = lane & 15;
  int kadd = (lane & 16) ? 16 : 0;
  const unsigned* q = (const unsigned*)p + ((m * ld + kadd) >> 1);
  v8u u;
#pragma unroll
  for (int v = 0; v < 8; ++v) u[v] = q[v];
  return __builtin_bit_cast(v16h, u);
}

// B fragment (32x16) where B = M^T, source row-major [N][K], ld even.
DEV v16h load_bt32(const _Float16* p, int ld) {
  int lane = threadIdx.x & 31;
  int n    = lane & 15;
  int koff = (lane & 16) ? 16 : 0;
  const unsigned* q = (const unsigned*)p + ((n * ld + koff) >> 1);
  v8u u;
#pragma unroll
  for (int v = 0; v < 8; ++v) u[v] = q[v];
  return __builtin_bit_cast(v16h, u);
}

// B fragment from K-pair-packed dword grid: dword (kp,n) = {B[2kp][n], B[2kp+1][n]}
DEV v16h load_bpk(const unsigned* q, int ldw) {
  int lane = threadIdx.x & 31;
  int n    = lane & 15;
  int kp8  = (lane & 16) ? 8 : 0;
  const unsigned* base = q + kp8 * ldw + n;
  v8u u;
#pragma unroll
  for (int v = 0; v < 8; ++v) u[v] = base[v * ldw];
  return __builtin_bit_cast(v16h, u);
}

// LDS byte offset of a __shared__ object (low 32 bits of its flat address)
DEV unsigned lds_off(const void* p) { return (unsigned)(size_t)p; }

// Async global->LDS b128 (GVS mode): per-lane 16B transfer, tracked on ASYNCcnt.
DEV void async_copy_b128(unsigned ldsaddr, unsigned voff, const void* sbase) {
  asm volatile("global_load_async_to_lds_b128 %0, %1, %2"
               :: "v"(ldsaddr), "v"(voff), "s"(sbase)
               : "memory");
}
DEV void wait_asynccnt0() { asm volatile("s_wait_asynccnt 0x0" ::: "memory"); }

// Problem: B=8, H=W=224, C=96, WS=7, NH=3, HD=32, N=49, 8192 windows.
static constexpr float kScale = 0.17677669529663687f;  // 32^-0.5

// ---------------------------------------------------------------------------
// Kernel 0a: weight fp32 -> f16, K-pair packed dwords [kp*N + n]
// ---------------------------------------------------------------------------
__global__ void convert_pack_kernel(const float* __restrict__ wa,
                                    const float* __restrict__ wb,
                                    const float* __restrict__ wp,
                                    unsigned* __restrict__ oa,
                                    unsigned* __restrict__ ob,
                                    unsigned* __restrict__ op) {
  int i = blockIdx.x * 256 + threadIdx.x;
  if (i < 48 * 288) {
    int kp = i / 288, n = i - kp * 288;
    union { _Float16 h[2]; unsigned u; } t;
    t.h[0] = (_Float16)wa[(2 * kp) * 288 + n];
    t.h[1] = (_Float16)wa[(2 * kp + 1) * 288 + n];
    oa[i] = t.u;
    t.h[0] = (_Float16)wb[(2 * kp) * 288 + n];
    t.h[1] = (_Float16)wb[(2 * kp + 1) * 288 + n];
    ob[i] = t.u;
  }
  if (i < 48 * 96) {
    int kp = i / 96, n = i - kp * 96;
    union { _Float16 h[2]; unsigned u; } t;
    t.h[0] = (_Float16)wp[(2 * kp) * 96 + n];
    t.h[1] = (_Float16)wp[(2 * kp + 1) * 96 + n];
    op[i] = t.u;
  }
}

// ---------------------------------------------------------------------------
// Kernel 0b: dense padded RPE table: rpeD[head][64][64] fp32, -1e30 in pads.
// ---------------------------------------------------------------------------
__global__ void build_rpe_kernel(const float* __restrict__ rpe_table,
                                 const int* __restrict__ rpe_index,
                                 float* __restrict__ rpeD) {
  int i = blockIdx.x * 256 + threadIdx.x;
  if (i >= 3 * 64 * 64) return;
  int head = i >> 12, r = (i >> 6) & 63, c = i & 63;
  float v = -1e30f;
  if (r < 49 && c < 49) v = rpe_table[head * 169 + rpe_index[r * 49 + c]];
  rpeD[i] = v;
}

// ---------------------------------------------------------------------------
// Kernel 1: LayerNorm + windowed QKV GEMM (templated on stream).
// STRM=0: rescaled -> K,V (cols 96..287 of qkv_a).
// STRM=1: rescaler -> Q (cols 0..95 of qkv_b, scaled).
// ---------------------------------------------------------------------------
template <int STRM>
__global__ __launch_bounds__(128) void qkv_kernel_t(
    const float* __restrict__ X, const float* __restrict__ g,
    const float* __restrict__ be, const unsigned* __restrict__ Wsrc,
    const float* __restrict__ qbias, _Float16* __restrict__ Qb,
    _Float16* __restrict__ Kb, _Float16* __restrict__ Vb) {
  constexpr int NCOLS  = STRM ? 96 : 192;
  constexpr int COLOFS = STRM ? 0 : 96;

  __shared__ __align__(16) _Float16 sX[64 * 96];
  __shared__ __align__(16) unsigned sW[48 * NCOLS];

  const int win = blockIdx.x;
  const int bb  = win >> 10;
  const int rem = win & 1023;
  const int wh  = rem >> 5, ww = rem & 31;
  const int tid = threadIdx.x, lane = tid & 31, wid = tid >> 5;

  // ---- LayerNorm (row = 96 elems, 3 per lane, wave32 xor-reduction) ----
  float gv0 = g[lane], gv1 = g[lane + 32], gv2 = g[lane + 64];
  float bv0 = be[lane], bv1 = be[lane + 32], bv2 = be[lane + 64];
  for (int r = wid; r < 49; r += 4) {
    int hh = wh * 7 + r / 7, wc = ww * 7 + r % 7;
    const float* src = X + ((size_t)((bb * 224 + hh) * 224 + wc)) * 96;
    float x0 = src[lane], x1 = src[lane + 32], x2 = src[lane + 64];
    float s = x0 + x1 + x2, s2 = x0 * x0 + x1 * x1 + x2 * x2;
#pragma unroll
    for (int o = 16; o; o >>= 1) {
      s  += __shfl_xor(s, o, 32);
      s2 += __shfl_xor(s2, o, 32);
    }
    float mu = s * (1.f / 96.f);
    float rs = rsqrtf(fmaxf(s2 * (1.f / 96.f) - mu * mu, 0.f) + 1e-5f);
    sX[r * 96 + lane]      = (_Float16)((x0 - mu) * rs * gv0 + bv0);
    sX[r * 96 + lane + 32] = (_Float16)((x1 - mu) * rs * gv1 + bv1);
    sX[r * 96 + lane + 64] = (_Float16)((x2 - mu) * rs * gv2 + bv2);
  }
  for (int i = tid; i < 15 * 48; i += 128) ((unsigned*)sX)[49 * 48 + i] = 0u;

  // ---- stage packed weight slice (dword copy; constexpr divisor) ----
  for (int i = tid; i < 48 * NCOLS; i += 128) {
    int kp = i / NCOLS, n = i - kp * NCOLS;
    sW[i] = Wsrc[kp * 288 + COLOFS + n];
  }
  __syncthreads();

  // ---- GEMM: [64x96] @ [96 x NCOLS], K in 3 steps of 32 ----
  const int mt = wid;
  const int n = lane & 15, mb = (lane >> 4) << 3;
  const int m0 = mt * 16 + mb;
  for (int nt = 0; nt < NCOLS / 16; ++nt) {
    v8f acc = {0.f, 0.f, 0.f, 0.f, 0.f, 0.f, 0.f, 0.f};
#pragma unroll
    for (int kk = 0; kk < 96; kk += 32) {
      v16h a = load_a32(sX + mt * 16 * 96 + kk, 96);
      v16h b = load_bpk(sW + (kk >> 1) * NCOLS + nt * 16, NCOLS);
      acc = wmma_f16(a, b, acc);
    }
    int gcol = COLOFS + nt * 16 + n;
    float bias = qbias[gcol];

    _Float16* buf;
    int c2;
    if constexpr (STRM == 1) {
      buf = Qb; c2 = gcol;
    } else {
      if (nt < 6) { buf = Kb; c2 = gcol - 96; }
      else        { buf = Vb; c2 = gcol - 192; }
    }
    _Float16* dst =
        buf + (unsigned)win * 4704u +
        (unsigned)(((unsigned)(c2 >> 5)) * 1568u + (unsigned)m0 * 32u +
                   (unsigned)(c2 & 31));
    if (m0 + 8 <= 49) {
#pragma unroll
      for (int r = 0; r < 8; ++r) {
        float val = acc[r] + bias;
        if constexpr (STRM == 1) val *= kScale;
        dst[r * 32] = (_Float16)val;
      }
    } else if (m0 < 49) {
#pragma unroll
      for (int r = 0; r < 8; ++r) {
        if (m0 + r < 49) {
          float val = acc[r] + bias;
          if constexpr (STRM == 1) val *= kScale;
          dst[r * 32] = (_Float16)val;
        }
      }
    }
  }
}

// ---------------------------------------------------------------------------
// Kernel 2: attention per (window, head). S = Q*K^T + rpeD; softmax; O = P*V.
// Q/K staged via async global->LDS (ASYNCcnt); V packed through VGPRs.
// ---------------------------------------------------------------------------
__global__ __launch_bounds__(128) void attn_kernel(
    const _Float16* __restrict__ Qb, const _Float16* __restrict__ Kb,
    const _Float16* __restrict__ Vb, const float* __restrict__ rpeD,
    _Float16* __restrict__ Ob) {
  __shared__ __align__(16) _Float16 sQ[64 * 32];   // rows 49..63 zero
  __shared__ __align__(16) _Float16 sK[64 * 32];   // rows 49..63 zero
  __shared__ __align__(16) unsigned sVp[32 * 32];  // K-pair packed V
  __shared__ __align__(16) float    sS[64 * 64];
  __shared__ __align__(16) _Float16 sP[64 * 64];

  const int win = blockIdx.x, head = blockIdx.y;
  const int tid = threadIdx.x, lane = tid & 31, wid = tid >> 5;

  const size_t base = ((size_t)win * 3 + head) * 49 * 32;
  __builtin_prefetch(Vb + base, 0, 0);            // global_prefetch_b8
  __builtin_prefetch(rpeD + head * 4096, 0, 0);

  // ---- async Q/K staging: 196 x b128 each (49 rows x 64B, 16B aligned) ----
  {
    const unsigned sqOff = lds_off(sQ), skOff = lds_off(sK);
    for (int i = tid; i < 196; i += 128) {
      unsigned voff = (unsigned)i * 16u;
      async_copy_b128(sqOff + voff, voff, Qb + base);
      async_copy_b128(skOff + voff, voff, Kb + base);
    }
  }
  for (int i = tid; i < 15 * 16; i += 128) {      // zero pad rows 49..63
    ((unsigned*)sQ)[784 + i] = 0u;
    ((unsigned*)sK)[784 + i] = 0u;
  }
  // V staged K-pair packed: dword (kp, d) = {V[2kp][d], V[2kp+1][d]}
  {
    const unsigned short* gv = (const unsigned short*)(Vb + base);
    for (int i = tid; i < 32 * 32; i += 128) {
      int kp = i >> 5, d = i & 31;
      int r0 = 2 * kp, r1 = 2 * kp + 1;
      unsigned lo = (r0 < 49) ? gv[r0 * 32 + d] : 0u;
      unsigned hi = (r1 < 49) ? gv[r1 * 32 + d] : 0u;
      sVp[i] = lo | (hi << 16);
    }
  }
  wait_asynccnt0();
  __syncthreads();

  const int n = lane & 15, mb = (lane >> 4) << 3;
  const int m0 = wid * 16 + mb;

  // ---- S = Q @ K^T + rpe (dense padded table: unconditional) ----
  {
    v16h a = load_a32(sQ + wid * 16 * 32, 32);
    for (int nt = 0; nt < 4; ++nt) {
      v8f acc = {0.f, 0.f, 0.f, 0.f, 0.f, 0.f, 0.f, 0.f};
      v16h b = load_bt32(sK + nt * 16 * 32, 32);
      acc = wmma_f16(a, b, acc);
      int j = nt * 16 + n;
      const float* rp = rpeD + head * 4096 + m0 * 64 + j;
      float* ss = sS + m0 * 64 + j;
#pragma unroll
      for (int r = 0; r < 8; ++r) ss[r * 64] = acc[r] + rp[r * 64];
    }
  }
  __syncthreads();

  // ---- softmax rows (fp32, float4 passes), write P in f16 quads (uint2) ----
  if (tid < 64) {
    float4* row4 = (float4*)(sS + tid * 64);
    float mx = -3e38f;
#pragma unroll 4
    for (int q4 = 0; q4 < 16; ++q4) {
      float4 v = row4[q4];
      mx = fmaxf(mx, fmaxf(fmaxf(v.x, v.y), fmaxf(v.z, v.w)));
    }
    float sum = 0.f;
#pragma unroll 4
    for (int q4 = 0; q4 < 16; ++q4) {
      float4 v = row4[q4];
      v.x = __expf(v.x - mx); v.y = __expf(v.y - mx);
      v.z = __expf(v.z - mx); v.w = __expf(v.w - mx);
      sum += (v.x + v.y) + (v.z + v.w);
      row4[q4] = v;
    }
    float inv = 1.f / sum;
    uint2* prow = (uint2*)(sP + tid * 64);
#pragma unroll 4
    for (int q4 = 0; q4 < 16; ++q4) {
      float4 v = row4[q4];
      union { _Float16 h[4]; uint2 u; } t;
      t.h[0] = (_Float16)(v.x * inv); t.h[1] = (_Float16)(v.y * inv);
      t.h[2] = (_Float16)(v.z * inv); t.h[3] = (_Float16)(v.w * inv);
      prow[q4] = t.u;
    }
  }
  __syncthreads();

  // ---- O = P @ V (K padded to 64; masked P columns are exactly 0) ----
  for (int nt = 0; nt < 2; ++nt) {
    v8f acc = {0.f, 0.f, 0.f, 0.f, 0.f, 0.f, 0.f, 0.f};
#pragma unroll
    for (int kk = 0; kk < 64; kk += 32) {
      v16h a = load_a32(sP + wid * 16 * 64 + kk, 64);
      v16h b = load_bpk(sVp + (kk >> 1) * 32 + nt * 16, 32);
      acc = wmma_f16(a, b, acc);
    }
    _Float16* dst = Ob + (unsigned)win * 4704u +
                    (unsigned)(m0 * 96 + head * 32 + nt * 16 + n);
    if (m0 + 8 <= 49) {
#pragma unroll
      for (int r = 0; r < 8; ++r) dst[r * 96] = (_Float16)acc[r];
    } else if (m0 < 49) {
#pragma unroll
      for (int r = 0; r < 8; ++r)
        if (m0 + r < 49) dst[r * 96] = (_Float16)acc[r];
    }
  }
}

// ---------------------------------------------------------------------------
// Kernel 3: output projection + window reverse. O and weights staged async.
// ---------------------------------------------------------------------------
__global__ __launch_bounds__(128) void proj_kernel(
    const _Float16* __restrict__ Ob, const unsigned* __restrict__ Wp,
    const float* __restrict__ pb, float* __restrict__ out) {
  __shared__ __align__(16) _Float16 sO[64 * 96];
  __shared__ __align__(16) unsigned sWp[48 * 96];  // K-pair packed weights

  const int win = blockIdx.x;
  const int bb = win >> 10, rem = win & 1023, wh = rem >> 5, ww = rem & 31;
  const int tid = threadIdx.x, lane = tid & 31, wid = tid >> 5;

  const _Float16* go = Ob + (size_t)win * 49 * 96;  // 9408B, 16B aligned
  {
    const unsigned soOff = lds_off(sO), swOff = lds_off(sWp);
    for (int i = tid; i < 588; i += 128) {  // 49*96 f16 = 588 x 16B
      unsigned voff = (unsigned)i * 16u;
      async_copy_b128(soOff + voff, voff, go);
    }
    for (int i = tid; i < 288; i += 128) {  // 288 x 16B = 1152 of the 48*96 dwords
      unsigned voff = (unsigned)i * 16u;
      async_copy_b128(swOff + voff, voff, Wp);
    }
  }
  for (int i = tid; i < 15 * 48; i += 128) ((unsigned*)sO)[2352 + i] = 0u;
  wait_asynccnt0();
  __syncthreads();

  const int mt = wid;
  const int n = lane & 15, mb = (lane >> 4) << 3;
  const int m0 = mt * 16 + mb;
  float* obase = out + ((size_t)((bb * 224 + wh * 7) * 224 + ww * 7)) * 96;

  for (int nt = 0; nt < 6; ++nt) {
    v8f acc = {0.f, 0.f, 0.f, 0.f, 0.f, 0.f, 0.f, 0.f};
#pragma unroll
    for (int kk = 0; kk < 96; kk += 32) {
      v16h a = load_a32(sO + mt * 16 * 96 + kk, 96);
      v16h b = load_bpk(sWp + (kk >> 1) * 96 + nt * 16, 96);
      acc = wmma_f16(a, b, acc);
    }
    int col = nt * 16 + n;
    float bias = pb[col];
    if (m0 + 8 <= 49) {
#pragma unroll
      for (int r = 0; r < 8; ++r) {
        int m = m0 + r, dh = m / 7, dw = m - dh * 7;
        obase[(unsigned)((dh * 224 + dw) * 96 + col)] = acc[r] + bias;
      }
    } else if (m0 < 49) {
#pragma unroll
      for (int r = 0; r < 8; ++r) {
        int m = m0 + r;
        if (m < 49) {
          int dh = m / 7, dw = m - dh * 7;
          obase[(unsigned)((dh * 224 + dw) * 96 + col)] = acc[r] + bias;
        }
      }
    }
  }
}

// ---------------------------------------------------------------------------
// Host launch
// ---------------------------------------------------------------------------
extern "C" void kernel_launch(void* const* d_in, const int* in_sizes, int n_in,
                              void* d_out, int out_size, void* d_ws,
                              size_t ws_size, hipStream_t stream) {
  (void)in_sizes; (void)n_in; (void)out_size; (void)ws_size;

  const float* rescaled  = (const float*)d_in[0];
  const float* rescaler  = (const float*)d_in[1];
  const float* norm_w_a  = (const float*)d_in[2];
  const float* norm_b_a  = (const float*)d_in[3];
  const float* norm_w_b  = (const float*)d_in[4];
  const float* norm_b_b  = (const float*)d_in[5];
  const float* qkv_w_a   = (const float*)d_in[6];
  const float* qkv_b_a   = (const float*)d_in[7];
  const float* qkv_w_b   = (const float*)d_in[8];
  const float* qkv_b_b   = (const float*)d_in[9];
  const float* rpe_table = (const float*)d_in[10];
  const float* proj_w    = (const float*)d_in[11];
  const float* proj_b    = (const float*)d_in[12];
  const int*   rpe_index = (const int*)d_in[13];
  float* out = (float*)d_out;

  char* ws = (char*)d_ws;
  size_t off = 0;
  auto alloc = [&](size_t bytes) -> void* {
    void* p = ws + off;
    off = (off + bytes + 255) & ~(size_t)255;
    return p;
  };
  unsigned* Wa16 = (unsigned*)alloc((size_t)48 * 288 * 4);
  unsigned* Wb16 = (unsigned*)alloc((size_t)48 * 288 * 4);
  unsigned* Wp16 = (unsigned*)alloc((size_t)48 * 96 * 4);
  float*    rpeD = (float*)alloc((size_t)3 * 64 * 64 * 4);
  _Float16* Qb   = (_Float16*)alloc((size_t)8192 * 3 * 49 * 32 * 2);
  _Float16* Kb   = (_Float16*)alloc((size_t)8192 * 3 * 49 * 32 * 2);
  _Float16* Vb   = (_Float16*)alloc((size_t)8192 * 3 * 49 * 32 * 2);
  _Float16* Ob   = (_Float16*)alloc((size_t)8192 * 49 * 96 * 2);

  convert_pack_kernel<<<dim3((48 * 288 + 255) / 256), dim3(256), 0, stream>>>(
      qkv_w_a, qkv_w_b, proj_w, Wa16, Wb16, Wp16);
  build_rpe_kernel<<<dim3((3 * 64 * 64 + 255) / 256), dim3(256), 0, stream>>>(
      rpe_table, rpe_index, rpeD);

  qkv_kernel_t<0><<<dim3(8192), dim3(128), 0, stream>>>(
      rescaled, norm_w_a, norm_b_a, Wa16, qkv_b_a, Qb, Kb, Vb);
  qkv_kernel_t<1><<<dim3(8192), dim3(128), 0, stream>>>(
      rescaler, norm_w_b, norm_b_b, Wb16, qkv_b_b, Qb, Kb, Vb);

  attn_kernel<<<dim3(8192, 3), dim3(128), 0, stream>>>(Qb, Kb, Vb, rpeD, Ob);

  proj_kernel<<<dim3(8192), dim3(128), 0, stream>>>(Ob, Wp16, proj_b, out);
}
